// When2comFusion_89069031784674
// MI455X (gfx1250) — compile-verified, matrix operand order checked by reading, not run on hardware
//
#include <hip/hip_runtime.h>
#include <hip/hip_bf16.h>
#include <math.h>

typedef __bf16 bf16_t;
typedef __attribute__((ext_vector_type(16))) __bf16 v16bf;
typedef __attribute__((ext_vector_type(8)))  float  v8f;

union FragU { v16bf v; uint4 q[2]; };

#define AGENTS 10
#define CH_IN  64
#define IMG_H  64
#define IMG_W  128

// ---------------------------------------------------------------------------
// 1) Affine warp: reference bilinear grid-sample. Writes fp32 NCHW `neighbor`
//    (for the final fusion) and bf16 zero-padded NHWC conv input.
// ---------------------------------------------------------------------------
__global__ void warp_kernel(const float* __restrict__ x, const float* __restrict__ T,
                            float* __restrict__ neighbor, bf16_t* __restrict__ in1P) {
  long e = (long)blockIdx.x * blockDim.x + threadIdx.x;
  const long total = (long)AGENTS * CH_IN * IMG_H * IMG_W;
  if (e >= total) return;
  int n   = (int)(e / (CH_IN * IMG_H * IMG_W));
  int rem = (int)(e % (CH_IN * IMG_H * IMG_W));
  int c   = rem / (IMG_H * IMG_W);
  int p   = rem % (IMG_H * IMG_W);
  int y   = p / IMG_W, xo = p % IMG_W;
  int b   = n / 5, j = n % 5;
  // t = pairwise_t_matrix[b,0,j,:2,[0,1,3]] with scalings
  const float* Tb = T + ((long)(b * 25 + j) * 16);
  float m00 = Tb[0];
  float m01 = Tb[1] * ((float)IMG_H / (float)IMG_W);
  float m02 = Tb[3] / (2.0f * 0.4f * (float)IMG_W) * 2.0f;
  float m10 = Tb[4] * ((float)IMG_W / (float)IMG_H);
  float m11 = Tb[5];
  float m12 = Tb[7] / (2.0f * 0.4f * (float)IMG_H) * 2.0f;
  float gx = -1.0f + 2.0f * (float)xo / (float)(IMG_W - 1);
  float gy = -1.0f + 2.0f * (float)y  / (float)(IMG_H - 1);
  float cx = m00 * gx + m01 * gy + m02;
  float cy = m10 * gx + m11 * gy + m12;
  float ix = (cx + 1.0f) * 0.5f * (float)(IMG_W - 1);
  float iy = (cy + 1.0f) * 0.5f * (float)(IMG_H - 1);
  float x0f = floorf(ix), y0f = floorf(iy);
  float wx = ix - x0f, wy = iy - y0f;
  int x0 = (int)x0f, y0 = (int)y0f;
  const float* src = x + ((long)n * CH_IN + c) * (IMG_H * IMG_W);
  auto samp = [&](int xi, int yi) -> float {
    if (xi < 0 || xi > IMG_W - 1 || yi < 0 || yi > IMG_H - 1) return 0.0f;
    return src[yi * IMG_W + xi];
  };
  float v = samp(x0, y0)     * (1.0f - wx) * (1.0f - wy)
          + samp(x0 + 1, y0) * wx          * (1.0f - wy)
          + samp(x0, y0 + 1) * (1.0f - wx) * wy
          + samp(x0 + 1, y0 + 1) * wx * wy;
  neighbor[e] = v;
  in1P[(((long)n * (IMG_H + 2) + (y + 1)) * (IMG_W + 2) + (xo + 1)) * CH_IN + c] = (bf16_t)v;
}

// ---------------------------------------------------------------------------
// 2) BN-fold + repack conv weights: OIHW fp32 -> [Cout][9][Cin] bf16, fused bias
// ---------------------------------------------------------------------------
__global__ void repack_conv(const float* __restrict__ w, const float* __restrict__ b,
                            const float* __restrict__ gamma, const float* __restrict__ beta,
                            const float* __restrict__ mean, const float* __restrict__ var,
                            bf16_t* __restrict__ wp, float* __restrict__ biasp,
                            int Cout, int Cin) {
  long idx = (long)blockIdx.x * blockDim.x + threadIdx.x;
  long total = (long)Cout * 9 * Cin;
  if (idx >= total) return;
  int cout = (int)(idx / (9 * Cin));
  int r    = (int)(idx % (9 * Cin));
  int kpos = r / Cin;
  int cin  = r % Cin;
  float scale = gamma[cout] * rsqrtf(var[cout] + 1e-5f);
  float wv = w[(long)(cout * Cin + cin) * 9 + kpos] * scale;
  wp[idx] = (bf16_t)wv;
  if (r == 0) biasp[cout] = (b[cout] - mean[cout]) * scale + beta[cout];
}

// ---------------------------------------------------------------------------
// 3) Implicit-GEMM conv3x3 (pad=1), bf16 WMMA, bias+ReLU epilogue.
//    Block tile 128(M pixels) x 64(N couts); 8 waves, each wave owns a 32x32
//    output tile = 2 A-frags x 2 B-frags -> 4 v_wmma per K-chunk.
//    Register double-buffering: next chunk's global loads issue before the
//    current chunk's WMMAs, hiding VMEM latency behind matrix math.
//    Padded NHWC activations -> zero predication around WMMA (EXEC all-ones).
// ---------------------------------------------------------------------------
__global__ void __launch_bounds__(256) conv_wmma(
    const bf16_t* __restrict__ inP, const bf16_t* __restrict__ wp,
    const float* __restrict__ bias, bf16_t* __restrict__ outP,
    int Hin, int Win, int Cin, int Hout, int Wout, int Cout,
    int stride, int outPad) {
  __shared__ bf16_t sA[128 * 32];   // 128 pixels x 32 K-halves (8 KB)
  __shared__ bf16_t sB[64 * 32];    // 64 couts x 32 K-halves (4 KB)
  const int tid    = threadIdx.x;
  const int lane   = tid & 31;
  const int waveId = tid >> 5;
  const int mw = waveId & 3;        // wave's 32-row band (0..3)
  const int nw = waveId >> 2;       // wave's 32-col band (0..1)
  const int m0 = blockIdx.x << 7;   // 128 pixels per block
  const int n0 = blockIdx.y << 6;   // 64 couts per block
  const int HWo  = Hout * Wout;
  const int WinP = Win + 2;

  // ---- staging roles ----
  // A: 128 rows x 64B; two threads per row, each stores 32B (2x uint4)
  const int aRow  = tid >> 1;
  const int aHalf = tid & 1;        // halves [aHalf*16 .. aHalf*16+15]
  // B: 64 rows x 64B; four threads per row, each stores 16B
  const int bRow = tid >> 2;
  const int bSeg = tid & 3;

  int am  = m0 + aRow;
  int img = am / HWo;
  int r   = am - img * HWo;
  int oy  = r / Wout;
  int ox  = r - oy * Wout;
  const long aRowBase = ((long)img * (Hin + 2)) * WinP;  // + iy*WinP + ix later
  const long bRowBase = (long)(n0 + bRow) * 9 * Cin;     // + kpos*Cin + cc later

  v8f acc[2][2];
  for (int i = 0; i < 8; ++i) {
    acc[0][0][i] = 0.0f; acc[0][1][i] = 0.0f;
    acc[1][0][i] = 0.0f; acc[1][1][i] = 0.0f;
  }

  const int lan15 = lane & 15;
  const int kb    = (lane >> 4) << 3;   // ISA bf16 operand swizzle: K 0..7/16..23 vs 8..15/24..31

  const int chunks = Cin >> 5;          // K chunks of 32 per filter tap
  const int totalIters = 9 * chunks;

  // prefetch registers for the first chunk (kpos=0, cc=0)
  uint4 pa0, pa1, pb;
  {
    int iy = oy * stride + 0, ix = ox * stride + 0;
    const bf16_t* aSrc = inP + (aRowBase + (long)iy * WinP + ix) * Cin + aHalf * 16;
    pa0 = *((const uint4*)aSrc);
    pa1 = *((const uint4*)(aSrc + 8));
    pb  = *((const uint4*)(wp + bRowBase + bSeg * 8));
  }

  int nkpos = 0, ncc = 0;  // (kpos, cc) of the chunk currently in pa/pb + advance state
  for (int it = 0; it < totalIters; ++it) {
    __syncthreads();  // previous chunk's readers done
    *((uint4*)&sA[aRow * 32 + aHalf * 16])     = pa0;
    *((uint4*)&sA[aRow * 32 + aHalf * 16 + 8]) = pa1;
    *((uint4*)&sB[bRow * 32 + bSeg * 8])       = pb;
    __syncthreads();  // staged data visible

    // issue next chunk's global loads before doing this chunk's math
    if (it + 1 < totalIters) {
      ncc += 32;
      if (ncc == Cin) { ncc = 0; ++nkpos; }
      int ky = nkpos / 3, kx = nkpos - ky * 3;
      int iy = oy * stride + ky, ix = ox * stride + kx;
      const bf16_t* aSrc = inP + (aRowBase + (long)iy * WinP + ix) * Cin + ncc + aHalf * 16;
      pa0 = *((const uint4*)aSrc);
      pa1 = *((const uint4*)(aSrc + 8));
      pb  = *((const uint4*)(wp + bRowBase + (long)nkpos * Cin + ncc + bSeg * 8));
    }

    FragU a0, a1, b0, b1;
    {
      int ar0 = ((mw << 5) + lan15) * 32;
      int ar1 = ((mw << 5) + 16 + lan15) * 32;
      a0.q[0] = *((const uint4*)&sA[ar0 + kb]);
      a0.q[1] = *((const uint4*)&sA[ar0 + kb + 16]);
      a1.q[0] = *((const uint4*)&sA[ar1 + kb]);
      a1.q[1] = *((const uint4*)&sA[ar1 + kb + 16]);
      int br0 = ((nw << 5) + lan15) * 32;
      int br1 = ((nw << 5) + 16 + lan15) * 32;
      b0.q[0] = *((const uint4*)&sB[br0 + kb]);
      b0.q[1] = *((const uint4*)&sB[br0 + kb + 16]);
      b1.q[0] = *((const uint4*)&sB[br1 + kb]);
      b1.q[1] = *((const uint4*)&sB[br1 + kb + 16]);
    }
    acc[0][0] = __builtin_amdgcn_wmma_f32_16x16x32_bf16(false, a0.v, false, b0.v,
                                                        (short)0, acc[0][0], false, false);
    acc[0][1] = __builtin_amdgcn_wmma_f32_16x16x32_bf16(false, a0.v, false, b1.v,
                                                        (short)0, acc[0][1], false, false);
    acc[1][0] = __builtin_amdgcn_wmma_f32_16x16x32_bf16(false, a1.v, false, b0.v,
                                                        (short)0, acc[1][0], false, false);
    acc[1][1] = __builtin_amdgcn_wmma_f32_16x16x32_bf16(false, a1.v, false, b1.v,
                                                        (short)0, acc[1][1], false, false);
  }

  // Epilogue: C layout lane l, vgpr v -> M = v + 8*(l>=16), N = l&15
  for (int i = 0; i < 2; ++i) {
    int mRowBase = m0 + (mw << 5) + (i << 4) + ((lane >> 4) << 3);
    for (int j = 0; j < 2; ++j) {
      int cout = n0 + (nw << 5) + (j << 4) + lan15;
      float bs = bias[cout];
      for (int v = 0; v < 8; ++v) {
        int mr = mRowBase + v;
        float val = acc[i][j][v] + bs;
        val = val > 0.0f ? val : 0.0f;
        int img2 = mr / HWo;
        int r2   = mr - img2 * HWo;
        int y2   = r2 / Wout;
        int x2   = r2 - y2 * Wout;
        long o;
        if (outPad)
          o = (((long)img2 * (Hout + 2) + y2 + 1) * (Wout + 2) + x2 + 1) * Cout + cout;
        else
          o = (((long)img2 * Hout + y2) * Wout + x2) * Cout + cout;
        outP[o] = (bf16_t)val;
      }
    }
  }
}

// ---------------------------------------------------------------------------
// 4) Big MLP layer 1 (131072 -> 256), weight-read bound: one pass over W,
//    all rows accumulated simultaneously, fp32 atomics into out[nRows][256].
//    feat is bf16 NHWC [agent][16][32][256]; reference flatten is NCHW.
// ---------------------------------------------------------------------------
__global__ void mlp_layer1(const bf16_t* __restrict__ feat, const float* __restrict__ Wm,
                           float* __restrict__ out, int rowStart, int rowStep, int nRows) {
  int o = threadIdx.x;           // output unit 0..255
  int slab = blockIdx.x;         // f-slab 0..255 (512 features each)
  float acc[10];
  for (int k = 0; k < 10; ++k) acc[k] = 0.0f;
  int f0 = slab * 512;
  for (int f = f0; f < f0 + 512; ++f) {
    float wv = Wm[(long)f * 256 + o];
    int cch = f >> 9;            // NCHW flat: f = c*512 + (y*32+x)
    int rr  = f & 511;
    long fi = (long)rr * 256 + cch;  // NHWC storage index
    for (int k = 0; k < nRows; ++k) {
      int ag = rowStart + k * rowStep;
      acc[k] += (float)feat[(long)ag * 131072 + fi] * wv;
    }
  }
  for (int k = 0; k < nRows; ++k) atomicAdd(&out[k * 256 + o], acc[k]);
}

// ---------------------------------------------------------------------------
// 5) Attention head: small MLP tails + attn + softmax -> watt[10]
// ---------------------------------------------------------------------------
__global__ void attention_head(
    const float* __restrict__ o1k, const float* __restrict__ o1q,
    const float* __restrict__ b1k,
    const float* __restrict__ Wk2, const float* __restrict__ bk2,
    const float* __restrict__ Wk3, const float* __restrict__ bk3,
    const float* __restrict__ b1q,
    const float* __restrict__ Wq2, const float* __restrict__ bq2,
    const float* __restrict__ Wq3, const float* __restrict__ bq3,
    const float* __restrict__ attW, const float* __restrict__ attB,
    float* __restrict__ watt) {
  __shared__ float h1k[10 * 256];
  __shared__ float h2k[10 * 128];
  __shared__ float keys[10 * 256];
  __shared__ float h1q[2 * 256];
  __shared__ float h2q[2 * 128];
  __shared__ float q3[2 * 32];
  __shared__ float qa[2 * 256];
  __shared__ float attn[10];
  int t = threadIdx.x;
  for (int i = t; i < 10 * 256; i += 256) {
    float v = o1k[i] + b1k[i & 255];
    h1k[i] = v > 0.0f ? v : 0.0f;
  }
  for (int i = t; i < 2 * 256; i += 256) {
    float v = o1q[i] + b1q[i & 255];
    h1q[i] = v > 0.0f ? v : 0.0f;
  }
  __syncthreads();
  for (int i = t; i < 10 * 128; i += 256) {
    int n = i >> 7, o = i & 127;
    float s = bk2[o];
    for (int kk = 0; kk < 256; ++kk) s += h1k[n * 256 + kk] * Wk2[kk * 128 + o];
    h2k[i] = s > 0.0f ? s : 0.0f;
  }
  for (int i = t; i < 2 * 128; i += 256) {
    int n = i >> 7, o = i & 127;
    float s = bq2[o];
    for (int kk = 0; kk < 256; ++kk) s += h1q[n * 256 + kk] * Wq2[kk * 128 + o];
    h2q[i] = s > 0.0f ? s : 0.0f;
  }
  __syncthreads();
  for (int i = t; i < 10 * 256; i += 256) {
    int n = i >> 8, o = i & 255;
    float s = bk3[o];
    for (int kk = 0; kk < 128; ++kk) s += h2k[n * 128 + kk] * Wk3[kk * 256 + o];
    keys[i] = s;
  }
  for (int i = t; i < 2 * 32; i += 256) {
    int n = i >> 5, o = i & 31;
    float s = bq3[o];
    for (int kk = 0; kk < 128; ++kk) s += h2q[n * 128 + kk] * Wq3[kk * 32 + o];
    q3[i] = s;
  }
  __syncthreads();
  for (int i = t; i < 2 * 256; i += 256) {
    int n = i >> 8, o = i & 255;
    float s = attB[o];
    for (int kk = 0; kk < 32; ++kk) s += q3[n * 32 + kk] * attW[kk * 256 + o];
    qa[i] = s;
  }
  __syncthreads();
  if (t < 10) {
    int b = t / 5;
    float s = 0.0f;
    for (int kk = 0; kk < 256; ++kk) s += keys[t * 256 + kk] * qa[b * 256 + kk];
    attn[t] = s;
  }
  __syncthreads();
  if (t < 2) {
    float mx = -1e30f;
    for (int j = 0; j < 5; ++j) mx = fmaxf(mx, attn[t * 5 + j]);
    float e[5], sum = 0.0f;
    for (int j = 0; j < 5; ++j) { e[j] = __expf(attn[t * 5 + j] - mx); sum += e[j]; }
    for (int j = 0; j < 5; ++j) watt[t * 5 + j] = e[j] / sum;
  }
}

// ---------------------------------------------------------------------------
// 6) Fusion: out[b] = sum_n watt[b*5+n] * neighbor[b*5+n]  (fp32 NCHW)
// ---------------------------------------------------------------------------
__global__ void fuse_kernel(const float* __restrict__ neighbor, const float* __restrict__ watt,
                            float* __restrict__ out) {
  long e = (long)blockIdx.x * blockDim.x + threadIdx.x;
  const long CHW = (long)CH_IN * IMG_H * IMG_W;
  if (e >= 2 * CHW) return;
  int b = (int)(e / CHW);
  long rem = e - (long)b * CHW;
  float s = 0.0f;
  for (int j = 0; j < 5; ++j)
    s += watt[b * 5 + j] * neighbor[(long)(b * 5 + j) * CHW + rem];
  out[e] = s;
}

// ---------------------------------------------------------------------------
extern "C" void kernel_launch(void* const* d_in, const int* in_sizes, int n_in,
                              void* d_out, int out_size, void* d_ws, size_t ws_size,
                              hipStream_t stream) {
  (void)in_sizes; (void)n_in; (void)out_size; (void)ws_size;
  const float* x = (const float*)d_in[0];
  const float* T = (const float*)d_in[2];
  // conv layer l params at d_in[3+6l .. 3+6l+5]: w,b,gamma,beta,mean,var
  // key_mlp: 33..38  query_mlp: 39..44  attn_W: 45  attn_b: 46

  char* ws = (char*)d_ws;
  size_t off = 0;
  auto take = [&](size_t bytes) -> char* {
    char* p = ws + off;
    off = (off + bytes + 255) & ~(size_t)255;
    return p;
  };
  // activations
  float*  neighbor = (float*) take((size_t)AGENTS * CH_IN * IMG_H * IMG_W * 4);       // 21 MB
  bf16_t* in1P  = (bf16_t*)take((size_t)AGENTS * 66 * 130 * 64  * 2);                 // 11 MB
  bf16_t* out1P = (bf16_t*)take((size_t)AGENTS * 66 * 130 * 512 * 2);                 // 88 MB
  bf16_t* out2P = (bf16_t*)take((size_t)AGENTS * 66 * 130 * 256 * 2);                 // 44 MB
  bf16_t* out3P = (bf16_t*)take((size_t)AGENTS * 34 * 66  * 256 * 2);                 // 11.5 MB
  bf16_t* out4P = (bf16_t*)take((size_t)AGENTS * 34 * 66  * 256 * 2);                 // 11.5 MB
  bf16_t* out5  = (bf16_t*)take((size_t)AGENTS * 16 * 32  * 256 * 2);                 // 2.6 MB
  // packed weights + fused biases
  bf16_t* wp[5]; float* bp[5];
  const int COUT[5] = {512, 256, 256, 256, 256};
  const int CIN[5]  = {64, 512, 256, 256, 256};
  for (int l = 0; l < 5; ++l) {
    wp[l] = (bf16_t*)take((size_t)COUT[l] * 9 * CIN[l] * 2);
    bp[l] = (float*) take((size_t)COUT[l] * 4);
  }
  float* o1k  = (float*)take(10 * 256 * 4);
  float* o1q  = (float*)take(2 * 256 * 4);
  float* watt = (float*)take(10 * 4);

  // zero padded halos + atomic accumulators (capturable memsets)
  hipMemsetAsync(in1P,  0, (size_t)AGENTS * 66 * 130 * 64  * 2, stream);
  hipMemsetAsync(out1P, 0, (size_t)AGENTS * 66 * 130 * 512 * 2, stream);
  hipMemsetAsync(out2P, 0, (size_t)AGENTS * 66 * 130 * 256 * 2, stream);
  hipMemsetAsync(out3P, 0, (size_t)AGENTS * 34 * 66  * 256 * 2, stream);
  hipMemsetAsync(out4P, 0, (size_t)AGENTS * 34 * 66  * 256 * 2, stream);
  hipMemsetAsync(o1k, 0, 10 * 256 * 4, stream);
  hipMemsetAsync(o1q, 0, 2 * 256 * 4, stream);

  // repack conv weights (BN folded)
  for (int l = 0; l < 5; ++l) {
    const float* w_  = (const float*)d_in[3 + 6 * l + 0];
    const float* b_  = (const float*)d_in[3 + 6 * l + 1];
    const float* g_  = (const float*)d_in[3 + 6 * l + 2];
    const float* be_ = (const float*)d_in[3 + 6 * l + 3];
    const float* mn_ = (const float*)d_in[3 + 6 * l + 4];
    const float* vr_ = (const float*)d_in[3 + 6 * l + 5];
    long tot = (long)COUT[l] * 9 * CIN[l];
    repack_conv<<<(int)((tot + 255) / 256), 256, 0, stream>>>(
        w_, b_, g_, be_, mn_, vr_, wp[l], bp[l], COUT[l], CIN[l]);
  }

  // warp
  {
    long tot = (long)AGENTS * CH_IN * IMG_H * IMG_W;
    warp_kernel<<<(int)((tot + 255) / 256), 256, 0, stream>>>(x, T, neighbor, in1P);
  }

  // conv chain (implicit GEMM, bf16 WMMA); M tiles of 128 pixels, N tiles of 64
  // L1: 64ch 64x128 -> 512ch 64x128
  conv_wmma<<<dim3(640, 8), 256, 0, stream>>>(in1P,  wp[0], bp[0], out1P,
      64, 128, 64, 64, 128, 512, 1, 1);
  // L2: 512 -> 256, 64x128
  conv_wmma<<<dim3(640, 4), 256, 0, stream>>>(out1P, wp[1], bp[1], out2P,
      64, 128, 512, 64, 128, 256, 1, 1);
  // L3: 256 -> 256, stride 2 -> 32x64
  conv_wmma<<<dim3(160, 4), 256, 0, stream>>>(out2P, wp[2], bp[2], out3P,
      64, 128, 256, 32, 64, 256, 2, 1);
  // L4: 256 -> 256, 32x64
  conv_wmma<<<dim3(160, 4), 256, 0, stream>>>(out3P, wp[3], bp[3], out4P,
      32, 64, 256, 32, 64, 256, 1, 1);
  // L5: 256 -> 256, stride 2 -> 16x32 (unpadded output)
  conv_wmma<<<dim3(40, 4), 256, 0, stream>>>(out4P, wp[4], bp[4], out5,
      32, 64, 256, 16, 32, 256, 2, 0);

  // big MLP layer-1: keys (all 10 agents) and query (agents 0 and 5)
  mlp_layer1<<<256, 256, 0, stream>>>(out5, (const float*)d_in[33], o1k, 0, 1, 10);
  mlp_layer1<<<256, 256, 0, stream>>>(out5, (const float*)d_in[39], o1q, 0, 5, 2);

  // attention head
  attention_head<<<1, 256, 0, stream>>>(
      o1k, o1q,
      (const float*)d_in[34], (const float*)d_in[35], (const float*)d_in[36],
      (const float*)d_in[37], (const float*)d_in[38],
      (const float*)d_in[40], (const float*)d_in[41], (const float*)d_in[42],
      (const float*)d_in[43], (const float*)d_in[44],
      (const float*)d_in[45], (const float*)d_in[46], watt);

  // fused weighted sum -> d_out (2,64,64,128) fp32
  {
    long tot = (long)2 * CH_IN * IMG_H * IMG_W;
    fuse_kernel<<<(int)((tot + 255) / 256), 256, 0, stream>>>(neighbor, watt, (float*)d_out);
  }
}
